// HetAgg_36077725287020
// MI455X (gfx1250) — matrix operand order repaired and checked by reading, not compiled, not run on hardware
//
#include <hip/hip_runtime.h>

#define EMBED_D 128
#define NUM_NODES 10000
#define NUM_EDGE_TYPES 14
#define BATCH_N 8192
#define EDGE_IN 1536   // 128*12
#define HET_IN  1792   // 128*14

typedef __attribute__((ext_vector_type(16))) __bf16 v16bf;
typedef __attribute__((ext_vector_type(8)))  float  v8f;

#define KC  128   // K-chunk staged per iteration (4 WMMA K-steps)
#define AS2 136   // LDS A row stride in bf16 (= 68 words: conflict-free for 16 rows, 16B aligned)

union Pk4 { __bf16 b[4]; uint2 u; };
union V16 { v16bf v; uint4 u[2]; };

// Tile GEMM: out = leakyrelu(A[rows] @ W + bias), W pre-split to bf16 hi/lo,
// stored column-major WT[n*K + k]. Block = (MT*16) rows x 128 cols, 8 waves;
// each wave owns 16 cols and MT 16-row accumulator subtiles (B reused MT times).
template<int MT>
__global__ __launch_bounds__(256) void het_gemm(
    const float* __restrict__ A, const int* __restrict__ rows,
    long long aStrideE, int lda, int K,
    const __bf16* __restrict__ WTh, const __bf16* __restrict__ WTl,
    long long wStrideE,
    const float* __restrict__ bias, int biasStrideE,
    float* __restrict__ out, int ldo, int colStrideE)
{
  __shared__ __bf16 AH[MT * 16 * AS2];   // hi plane, natural K layout (pos = k)
  __shared__ __bf16 AL[MT * 16 * AS2];   // lo plane

  const int t    = threadIdx.x;
  const int wave = t >> 5;
  const int lane = t & 31;
  const int half = lane >> 4;
  const int nl   = lane & 15;
  const int n0   = wave << 4;
  const int tileM = blockIdx.x;
  const int e     = blockIdx.y;

  // staging geometry: thread owns float4 K-column `seg` of rows (t>>5) + 8*i
  const int seg  = (t & 31) << 2;                         // 0..124
  const int kq   = seg & 31;
  const int g    = kq >> 3;
  const int pg   = g;                                     // group order 0,1,2,3
  const int perm = (seg & ~31) + (pg << 3) + (kq & 7);    // pos within row (equals seg)

  int srcRowIdx[2 * MT];
#pragma unroll
  for (int i = 0; i < 2 * MT; ++i) {
    const int gRow = tileM * (MT * 16) + (t >> 5) + 8 * i;
    srcRowIdx[i] = rows ? rows[gRow] : gRow;              // gather or identity
  }
  const float* Abase = A + (long long)e * aStrideE + seg;

  const __bf16* bhRow = WTh + (long long)e * wStrideE + (long long)(n0 + nl) * K;
  const __bf16* blRow = WTl + (long long)e * wStrideE + (long long)(n0 + nl) * K;

  v8f acc[MT] = {};

  for (int k0 = 0; k0 < K; k0 += KC) {
    __syncthreads();
#pragma unroll
    for (int i = 0; i < 2 * MT; ++i) {
      // coalesced f32 gather load, split to bf16 hi/lo (conflict-free uint2 stores)
      const float4 v = *(const float4*)(Abase + (long long)srcRowIdx[i] * lda + k0);
      Pk4 h4, l4;
      float v0 = v.x, v1 = v.y, v2 = v.z, v3 = v.w;
      h4.b[0] = (__bf16)v0; l4.b[0] = (__bf16)(v0 - (float)h4.b[0]);
      h4.b[1] = (__bf16)v1; l4.b[1] = (__bf16)(v1 - (float)h4.b[1]);
      h4.b[2] = (__bf16)v2; l4.b[2] = (__bf16)(v2 - (float)h4.b[2]);
      h4.b[3] = (__bf16)v3; l4.b[3] = (__bf16)(v3 - (float)h4.b[3]);
      const int r = (t >> 5) + 8 * i;
      *(uint2*)&AH[r * AS2 + perm] = h4.u;
      *(uint2*)&AL[r * AS2 + perm] = l4.u;
    }
    __syncthreads();

#pragma unroll
    for (int blk = 0; blk < KC / 32; ++blk) {
      const int kb = blk * 32 + 16 * half;
      // B fragment: N = nl, K contiguous from k0+blk*32+16h (2x global_load_b128, L2-hot)
      const v16bf bh = *(const v16bf*)(bhRow + k0 + kb);
      const v16bf bl = *(const v16bf*)(blRow + k0 + kb);
#pragma unroll
      for (int ms = 0; ms < MT; ++ms) {
        // A fragment: M = ms*16+nl; two 16B halves (bank-conflict-free stride)
        const __bf16* pa = &AH[(ms * 16 + nl) * AS2 + kb];
        const __bf16* pl = &AL[(ms * 16 + nl) * AS2 + kb];
        V16 ah, al;
        ah.u[0] = *(const uint4*)pa;  ah.u[1] = *(const uint4*)(pa + 8);
        al.u[0] = *(const uint4*)pl;  al.u[1] = *(const uint4*)(pl + 8);
        // split-precision: hi*hi + hi*lo + lo*hi
        acc[ms] = __builtin_amdgcn_wmma_f32_16x16x32_bf16(false, ah.v, false, bh, (short)0, acc[ms], false, false);
        acc[ms] = __builtin_amdgcn_wmma_f32_16x16x32_bf16(false, ah.v, false, bl, (short)0, acc[ms], false, false);
        acc[ms] = __builtin_amdgcn_wmma_f32_16x16x32_bf16(false, al.v, false, bh, (short)0, acc[ms], false, false);
      }
    }
  }

  // epilogue: bias + leaky_relu(0.01); C/D layout: VGPR r -> row r + 8*half, col n0+nl
  const int   colOff = e * colStrideE;
  const float bv = bias[e * biasStrideE + n0 + nl];
#pragma unroll
  for (int ms = 0; ms < MT; ++ms) {
#pragma unroll
    for (int r = 0; r < 8; ++r) {
      float x = acc[ms][r] + bv;
      x = (x >= 0.0f) ? x : 0.01f * x;
      out[(long long)(tileM * (MT * 16) + ms * 16 + 8 * half + r) * ldo + colOff + n0 + nl] = x;
    }
  }
}

// One-time weight split+transpose: W[K x 128] f32 -> WT_hi/WT_lo[n*K + k] bf16
__global__ __launch_bounds__(256) void wsplit(
    const float* __restrict__ W, __bf16* __restrict__ Th, __bf16* __restrict__ Tl, int K)
{
  const long long e = blockIdx.y;
  const float* We  = W  + e * (long long)K * EMBED_D;
  __bf16*      The = Th + e * (long long)K * EMBED_D;
  __bf16*      Tle = Tl + e * (long long)K * EMBED_D;
  const int idx = blockIdx.x * 256 + threadIdx.x;   // over K*128
  const int n = idx & (EMBED_D - 1);
  const int k = idx >> 7;
  const float x = We[(long long)k * EMBED_D + n];
  const __bf16 h = (__bf16)x;
  The[(long long)n * K + k] = h;
  Tle[(long long)n * K + k] = (__bf16)(x - (float)h);
}

extern "C" void kernel_launch(void* const* d_in, const int* in_sizes, int n_in,
                              void* d_out, int out_size, void* d_ws, size_t ws_size,
                              hipStream_t stream) {
  (void)in_sizes; (void)n_in; (void)out_size; (void)ws_size;
  const float* features = (const float*)d_in[0];
  const float* W_edge   = (const float*)d_in[1];
  const float* b_edge   = (const float*)d_in[2];
  const float* W_het    = (const float*)d_in[3];
  const float* b_het    = (const float*)d_in[4];
  const int*   gid      = (const int*)d_in[5];
  float* outp = (float*)d_out;

  // workspace layout
  char* ws = (char*)d_ws;
  float* agg = (float*)ws;                                   // 8192 x 1792 f32 (~58.7 MB)
  size_t off = (size_t)BATCH_N * HET_IN * sizeof(float);
  __bf16* WeH = (__bf16*)(ws + off); off += (size_t)NUM_EDGE_TYPES * EDGE_IN * EMBED_D * 2;
  __bf16* WeL = (__bf16*)(ws + off); off += (size_t)NUM_EDGE_TYPES * EDGE_IN * EMBED_D * 2;
  __bf16* WhH = (__bf16*)(ws + off); off += (size_t)HET_IN * EMBED_D * 2;
  __bf16* WhL = (__bf16*)(ws + off);

  // split/transpose weights once (tiny; L2-resident afterwards)
  wsplit<<<dim3((EDGE_IN * EMBED_D) / 256, NUM_EDGE_TYPES), 256, 0, stream>>>(W_edge, WeH, WeL, EDGE_IN);
  wsplit<<<dim3((HET_IN  * EMBED_D) / 256, 1),              256, 0, stream>>>(W_het,  WhH, WhL, HET_IN);

  // stage 1: agg[b, e*128+n] = leakyrelu(features[e, gid[b], :] @ W_edge[e] + b_edge[e])
  // 64-row M-tiles (MT=4): 4x B-fragment reuse -> L2 weight traffic ~1.4 GB instead of 5.6 GB
  het_gemm<4><<<dim3(BATCH_N / 64, NUM_EDGE_TYPES), 256, 0, stream>>>(
      features, gid, (long long)NUM_NODES * EDGE_IN, EDGE_IN, EDGE_IN,
      WeH, WeL, (long long)EDGE_IN * EMBED_D,
      b_edge, EMBED_D,
      agg, HET_IN, EMBED_D);

  // stage 2: out = leakyrelu(agg @ W_het + b_het); MT=1 keeps 512 blocks for occupancy
  het_gemm<1><<<dim3(BATCH_N / 16, 1), 256, 0, stream>>>(
      agg, nullptr, 0, HET_IN, HET_IN,
      WhH, WhL, 0,
      b_het, 0,
      outp, EMBED_D, 0);
}
